// DeepSpeedMLP_6313601925530
// MI455X (gfx1250) — compile-verified, matrix operand order checked by reading, not run on hardware
//
#include <hip/hip_runtime.h>
#include <hip/hip_bf16.h>

// ---------------------------------------------------------------------------
// DeepSpeed MLP block: residual+bias add -> LayerNorm -> GEMM1+GeLU -> GEMM2
// -> +residual_add +output_b.   B=4,S=2048 -> M=8192 tokens, H=2048, I=8192.
// Compute-bound (~550 GFLOP vs ~0.5GB traffic) -> bf16 WMMA w/ fp32 accum.
// Staging uses gfx1250 GLOBAL_LOAD_ASYNC_TO_LDS_B128 + ASYNCcnt double-buffer.
// ---------------------------------------------------------------------------

typedef __attribute__((ext_vector_type(16))) __bf16 v16bf;
typedef __attribute__((ext_vector_type(8)))  float  v8f;

#define MTOT 8192
#define HDIM 2048
#define IDIM 8192

// round-to-nearest-even float -> bf16 (bit trick)
__device__ __forceinline__ unsigned short f2bf_bits(float f) {
  union { float f; unsigned u; } v; v.f = f;
  unsigned r = v.u + 0x7FFFu + ((v.u >> 16) & 1u);
  return (unsigned short)(r >> 16);
}
__device__ __forceinline__ __bf16 f2bf(float f) {
  unsigned short b = f2bf_bits(f);
  return __builtin_bit_cast(__bf16, b);
}

// flat address of a __shared__ object -> 32-bit LDS byte offset
__device__ __forceinline__ unsigned lds_off(const void* p) {
  return (unsigned)(unsigned long long)(uintptr_t)p;
}

// gfx1250 async DMA: global -> LDS, 16B per lane, tracked by ASYNCcnt
__device__ __forceinline__ void async_b128(unsigned lds_byte, const void* gaddr) {
  asm volatile("global_load_async_to_lds_b128 %0, %1, off"
               :: "v"(lds_byte), "v"(gaddr) : "memory");
}
__device__ __forceinline__ void wait_async0() {
  asm volatile("s_wait_asynccnt 0x0" ::: "memory");
}

// ---------------------------------------------------------------------------
// Kernel 1: fp32 [R][C] -> bf16 [C][R]  (weights become K-contiguous per column)
// ---------------------------------------------------------------------------
__global__ void __launch_bounds__(256)
transpose_cvt(const float* __restrict__ src, __bf16* __restrict__ dst,
              int R, int C) {
  __shared__ float t[32][33];
  const int tx = threadIdx.x, ty = threadIdx.y;        // 32 x 8
  const int c0 = blockIdx.x * 32, r0 = blockIdx.y * 32;
#pragma unroll
  for (int i = 0; i < 4; ++i)
    t[ty + i * 8][tx] = src[(size_t)(r0 + ty + i * 8) * C + c0 + tx];
  __syncthreads();
#pragma unroll
  for (int i = 0; i < 4; ++i)
    dst[(size_t)(c0 + ty + i * 8) * R + r0 + tx] = f2bf(t[tx][ty + i * 8]);
}

// ---------------------------------------------------------------------------
// Kernel 2: residual_add = input + residual + bias ; LayerNorm -> bf16
// One block (256 thr) per row of H=2048; 8 elements per thread, vectorized.
// ---------------------------------------------------------------------------
__global__ void __launch_bounds__(256)
ln_kernel(const float* __restrict__ in, const float* __restrict__ resid,
          const float* __restrict__ bias, const float* __restrict__ gamma,
          const float* __restrict__ beta, float* __restrict__ res_add,
          __bf16* __restrict__ lnout) {
  __shared__ float red[512];
  const int row = blockIdx.x;
  const int tid = threadIdx.x;
  const size_t base = (size_t)row * HDIM;

  float4 v[2];
  float s = 0.f, s2 = 0.f;
#pragma unroll
  for (int i = 0; i < 2; ++i) {
    const int off = tid * 4 + i * 1024;
    float4 a = *(const float4*)(in + base + off);
    float4 b = *(const float4*)(resid + base + off);
    float4 c = *(const float4*)(bias + off);
    float4 t;
    t.x = a.x + b.x + c.x;  t.y = a.y + b.y + c.y;
    t.z = a.z + b.z + c.z;  t.w = a.w + b.w + c.w;
    v[i] = t;
    *(float4*)(res_add + base + off) = t;
    s  += t.x + t.y + t.z + t.w;
    s2 += t.x * t.x + t.y * t.y + t.z * t.z + t.w * t.w;
  }
  red[tid] = s; red[tid + 256] = s2;
  __syncthreads();
  for (int st = 128; st > 0; st >>= 1) {
    if (tid < st) { red[tid] += red[tid + st]; red[tid + 256] += red[tid + 256 + st]; }
    __syncthreads();
  }
  const float mu   = red[0]   * (1.0f / HDIM);
  const float var  = red[256] * (1.0f / HDIM) - mu * mu;
  const float rstd = rsqrtf(var + 1e-6f);

#pragma unroll
  for (int i = 0; i < 2; ++i) {
    const int off = tid * 4 + i * 1024;
    float4 g = *(const float4*)(gamma + off);
    float4 b = *(const float4*)(beta + off);
    ushort4 o;
    o.x = f2bf_bits((v[i].x - mu) * rstd * g.x + b.x);
    o.y = f2bf_bits((v[i].y - mu) * rstd * g.y + b.y);
    o.z = f2bf_bits((v[i].z - mu) * rstd * g.z + b.z);
    o.w = f2bf_bits((v[i].w - mu) * rstd * g.w + b.w);
    *(ushort4*)(lnout + base + off) = o;
  }
}

// ---------------------------------------------------------------------------
// WMMA bf16 GEMM:  C[M][N] = A[M][K] * Bt[N][K]^T   (Bt is K-contiguous/col)
// 256 thr = 8 wave32; block tile 128x128, BK=64; wave tile 32x64 (2x4 WMMA).
// Double-buffered LDS fed by global_load_async_to_lds_b128 (ASYNCcnt).
// EPI==0: + bias, exact GeLU, bf16 out.  EPI==1: + resid + bias, f32 out.
// ---------------------------------------------------------------------------
template <int KDIM, int NDIM, int EPI>
__global__ void __launch_bounds__(256)
gemm_bf16(const __bf16* __restrict__ A, const __bf16* __restrict__ Bt,
          const float* __restrict__ bias, const float* __restrict__ resid,
          float* __restrict__ outF, __bf16* __restrict__ outB) {
  constexpr int BK = 64;
  constexpr int LD = BK + 8;           // 144B row pitch: breaks bank conflicts
  constexpr int TILE = 128 * LD;       // elements per buffer
  constexpr unsigned TILEB = TILE * 2; // bytes per buffer
  __shared__ __attribute__((aligned(16))) __bf16 As[2 * TILE];
  __shared__ __attribute__((aligned(16))) __bf16 Bs[2 * TILE];

  const int tid  = threadIdx.x;
  const int lane = tid & 31;
  const int w    = tid >> 5;
  const int wm   = (w & 3) * 32;   // wave M offset in block tile
  const int wn   = (w >> 2) * 64;  // wave N offset in block tile
  const int row0 = blockIdx.y * 128;
  const int col0 = blockIdx.x * 128;

  // cooperative tile-load indices: 16B (8 bf16) per thread per pass
  const int lr = tid >> 3;          // 0..31
  const int lc = (tid & 7) * 8;     // 0..56

  // per-thread global bases (elements) and LDS byte bases (buffer 0)
  const __bf16* gA = A  + (size_t)(row0 + lr) * KDIM + lc;
  const __bf16* gB = Bt + (size_t)(col0 + lr) * KDIM + lc;
  const unsigned la0 = lds_off(&As[lr * LD + lc]);
  const unsigned lb0 = lds_off(&Bs[lr * LD + lc]);

  // ISA 16-bit A-operand layout: lane<16 -> K {0..7,16..23}; lane>=16 -> +8
  const int fm = lane & 15;
  const int fk = (lane >> 4) * 8;

  v8f acc[2][4] = {};

  // stage tile at K offset k0 into buffer `buf` (async DMA, no VGPR bounce)
  auto stage = [&](int k0, int buf) {
    const unsigned lb_add = (unsigned)buf * TILEB;
#pragma unroll
    for (int p = 0; p < 4; ++p) {
      async_b128(la0 + lb_add + (unsigned)(p * 32 * LD * 2),
                 gA + (size_t)k0 + (size_t)p * 32 * KDIM);
      async_b128(lb0 + lb_add + (unsigned)(p * 32 * LD * 2),
                 gB + (size_t)k0 + (size_t)p * 32 * KDIM);
    }
  };

  stage(0, 0);
  wait_async0();
  __syncthreads();

  int buf = 0;
  for (int k0 = 0; k0 < KDIM; k0 += BK) {
    if (k0 + BK < KDIM) stage(k0 + BK, buf ^ 1);  // DMA next slab during math

    const __bf16* Ab = &As[buf * TILE];
    const __bf16* Bb = &Bs[buf * TILE];
#pragma unroll
    for (int ks = 0; ks < 2; ++ks) {              // two K=32 WMMA steps
      v16bf af[2], bf[4];
#pragma unroll
      for (int mi = 0; mi < 2; ++mi) {
        const __bf16* p = &Ab[(wm + mi * 16 + fm) * LD + ks * 32 + fk];
        union { v16bf v; uint4 q[2]; } u;
        u.q[0] = *(const uint4*)(p);
        u.q[1] = *(const uint4*)(p + 16);
        af[mi] = u.v;
      }
#pragma unroll
      for (int ni = 0; ni < 4; ++ni) {
        const __bf16* p = &Bb[(wn + ni * 16 + fm) * LD + ks * 32 + fk];
        union { v16bf v; uint4 q[2]; } u;
        u.q[0] = *(const uint4*)(p);
        u.q[1] = *(const uint4*)(p + 16);
        bf[ni] = u.v;
      }
#pragma unroll
      for (int mi = 0; mi < 2; ++mi)
#pragma unroll
        for (int ni = 0; ni < 4; ++ni)
          acc[mi][ni] = __builtin_amdgcn_wmma_f32_16x16x32_bf16(
              false, af[mi], false, bf[ni], (short)0, acc[mi][ni], false, false);
    }
    wait_async0();      // next slab landed in LDS
    __syncthreads();    // everyone done reading current slab
    buf ^= 1;
  }

  // C/D layout: VGPR j, lane l -> row = j + 8*(l>=16), col = l&15
  const int lh  = (lane >> 4) * 8;
  const int lc2 = lane & 15;
#pragma unroll
  for (int mi = 0; mi < 2; ++mi) {
#pragma unroll
    for (int ni = 0; ni < 4; ++ni) {
#pragma unroll
      for (int j = 0; j < 8; ++j) {
        const int r = row0 + wm + mi * 16 + j + lh;
        const int c = col0 + wn + ni * 16 + lc2;
        float x = acc[mi][ni][j];
        if (EPI == 0) {
          x += bias[c];
          const float g = 0.5f * x * (1.0f + erff(x * 0.70710678118654752f));
          outB[(size_t)r * NDIM + c] = f2bf(g);
        } else {
          x += resid[(size_t)r * NDIM + c] + bias[c];
          outF[(size_t)r * NDIM + c] = x;
        }
      }
    }
  }
}

// ---------------------------------------------------------------------------
extern "C" void kernel_launch(void* const* d_in, const int* in_sizes, int n_in,
                              void* d_out, int out_size, void* d_ws, size_t ws_size,
                              hipStream_t stream) {
  const float* input    = (const float*)d_in[0];
  const float* residual = (const float*)d_in[1];
  // d_in[2] residual_norm: unused by the reference
  const float* bias     = (const float*)d_in[3];
  const float* gamma    = (const float*)d_in[4];
  const float* beta     = (const float*)d_in[5];
  const float* inter_w  = (const float*)d_in[6];
  const float* inter_b  = (const float*)d_in[7];
  const float* output_w = (const float*)d_in[8];
  const float* output_b = (const float*)d_in[9];
  float* out = (float*)d_out;

  char* ws = (char*)d_ws;
  float*  res_add = (float*) (ws);                          //  64 MiB [M][H] f32
  __bf16* ln      = (__bf16*)(ws + ((size_t)64  << 20));    //  32 MiB [M][H] bf16
  __bf16* w1t     = (__bf16*)(ws + ((size_t)96  << 20));    //  32 MiB [I][H] bf16
  __bf16* w2t     = (__bf16*)(ws + ((size_t)128 << 20));    //  32 MiB [H][I] bf16
  __bf16* hbuf    = (__bf16*)(ws + ((size_t)160 << 20));    // 128 MiB [M][I] bf16

  dim3 tb(32, 8);
  transpose_cvt<<<dim3(IDIM / 32, HDIM / 32), tb, 0, stream>>>(inter_w,  w1t, HDIM, IDIM);
  transpose_cvt<<<dim3(HDIM / 32, IDIM / 32), tb, 0, stream>>>(output_w, w2t, IDIM, HDIM);

  ln_kernel<<<MTOT, 256, 0, stream>>>(input, residual, bias, gamma, beta,
                                      res_add, ln);

  gemm_bf16<HDIM, IDIM, 0><<<dim3(IDIM / 128, MTOT / 128), 256, 0, stream>>>(
      ln, w1t, inter_b, nullptr, nullptr, hbuf);

  gemm_bf16<IDIM, HDIM, 1><<<dim3(HDIM / 128, MTOT / 128), 256, 0, stream>>>(
      hbuf, w2t, output_b, res_add, out, nullptr);
}
